// ReptileBrain_15195594293744
// MI455X (gfx1250) — matrix operand order repaired and checked by reading, not compiled, not run on hardware
//
#include <hip/hip_runtime.h>

#define T_STEPS 512
#define BATCH   128
#define HID     256
#define IN_D    8
#define SW      256      // state/res row width (floats)
#define ABW     256      // A-staging row width (halfs)
#define NFRAGS  338      // total packed 32x16 B-tiles

typedef _Float16 f16;
typedef __attribute__((ext_vector_type(16))) _Float16 v16h;
typedef __attribute__((ext_vector_type(8)))  float    v8f;
typedef __attribute__((ext_vector_type(4)))  float    f32x4;

// Regions in ORDER: olf, opt, bs, tec, stri, pal, cer, hyp, spi, mot
constexpr int   R_H[10]   = {20,17,38,38,30,25,25,20,20,17};
constexpr int   R_OFF[10] = {0,20,37,75,113,143,168,193,213,233};   // offsets in 250-vec
constexpr float R_DT[10]  = {1.5f,1.5f,2.0f,1.8f,1.0f,0.4f,1.0f,0.8f,3.0f,1.2f};

// Output buffer offsets (flat f32, tuple order: preds, hidden, anomaly, surprise, props)
constexpr long O_HIDDEN = (long)BATCH*(T_STEPS-1)*IN_D;
constexpr long O_ANOM   = O_HIDDEN + (long)BATCH*T_STEPS*HID;
constexpr long O_SURP   = O_ANOM   + (long)BATCH*T_STEPS;
constexpr long O_PROPS  = O_SURP   + (long)BATCH*(T_STEPS-1);

// ---------------- packed-weight prep ------------------------------------
struct MatDesc { const float* W0; const float* W1; int K0, K, N, fragBase, Ktiles, Ntiles; };
struct PackArgs { MatDesc m[16]; };

// One block per fragment (32x16 B-tile of W^T). Lane l (=tid>>4) supplies column
// n = nt*16 + (l&15); K rows 16*(l>>4) + idx, halfs stored contiguously so the scan
// kernel loads each fragment as 2x global_load_b128 per lane.
__global__ __launch_bounds__(512) void pack_kernel(PackArgs pa, f16* pack) {
  int f = blockIdx.x;
  int mi = 0;
#pragma unroll
  for (int i = 1; i < 16; ++i) if (f >= pa.m[i].fragBase) mi = i;
  MatDesc d = pa.m[mi];
  int local = f - d.fragBase;
  int nt = local / d.Ktiles, kt = local - nt * d.Ktiles;
  int tid = threadIdx.x;
  int lane = tid >> 4, hx = tid & 15;
  int n = nt*16 + (lane & 15);
  int k = kt*32 + (lane >> 4)*16 + hx;
  float v = 0.f;
  if (n < d.N && k < d.K)
    v = (k < d.K0) ? d.W0[n*d.K0 + k] : d.W1[n*(d.K - d.K0) + (k - d.K0)];
  pack[(long)f*512 + lane*16 + hx] = (f16)v;
}

// ---------------- WMMA fragment loads -----------------------------------
// A (16x32 f16): lane m=l&15, half=l>>4; element s -> k=16*half+s
__device__ __forceinline__ v16h load_a(const f16* aB, int kCol, int lane) {
  int m = lane & 15, hf = lane >> 4;
  const f16* p = aB + m*ABW + kCol + hf*16;
  union { f32x4 q[2]; v16h h; } u;
  u.q[0] = *(const f32x4*)(p);
  u.q[1] = *(const f32x4*)(p + 8);
  return u.h;
}
// B: packed contiguously per lane in d_ws
__device__ __forceinline__ v16h load_b(const f16* pack, int frag, int lane) {
  const f16* p = pack + (long)frag*512 + lane*16;
  union { f32x4 q[2]; v16h h; } u;
  u.q[0] = *(const f32x4*)(p);
  u.q[1] = *(const f32x4*)(p + 8);
  return u.h;
}

// block-cooperative M=16 GEMM: res[16 x Nt*16] (+colBase) = aBuf[16 x Kt*32] @ packed
__device__ __forceinline__ void gemm16(float* res, int colBase,
                                       const f16* aB, int aCol, const f16* pack,
                                       int fragBase, int Kt, int Nt, int wave, int lane) {
  for (int nt = wave; nt < Nt; nt += 8) {
    if (nt + 8 < Nt) __builtin_prefetch(pack + (long)(fragBase + (nt+8)*Kt)*512, 0, 1);
    v8f acc = {};
    for (int kt = 0; kt < Kt; ++kt) {
      v16h a = load_a(aB, aCol + kt*32, lane);
      v16h b = load_b(pack, fragBase + nt*Kt + kt, lane);
      acc = __builtin_amdgcn_wmma_f32_16x16x32_f16(false, a, false, b, (short)0, acc, false, false);
    }
    int n  = colBase + nt*16 + (lane & 15);
    int mh = (lane >> 4) * 8;
#pragma unroll
    for (int r = 0; r < 8; ++r) res[(mh + r)*SW + n] = acc[r];
  }
}

// ---------------- scan kernel -------------------------------------------
struct ScanArgs {
  const float* x; const float* metab;
  const float* inb0; const float* inb1;
  const float* bi[10]; const float* bh[10]; const float* ls[10];
  const float* g[10];  const float* be[10];
  const float* p2b_pb; const float* p2b_gb;
  const float* th_b; const float* th_g; const float* th_b2; const float* tp_b;
  const f16* pack;
  float* hid;
};

template<int RI>
__device__ __forceinline__ void region_ew(int slot, int resCol, float* sN, const float* sO,
                                          const float* res, const ScanArgs& A, float mdt, int tid) {
  int r = tid - slot*16;
  if (r >= 0 && r < 16) {
    constexpr int H_ = R_H[RI];
    constexpr int OFF_ = R_OFF[RI];
    const float dts = R_DT[RI];
    const float *bi = A.bi[RI], *bh = A.bh[RI], *ls = A.ls[RI], *g = A.g[RI], *be = A.be[RI];
    float hn[H_];
    float mean = 0.f;
#pragma unroll
    for (int j = 0; j < H_; ++j) {
      float cand = tanhf(res[r*SW + resCol + j] + bi[j] + bh[j]);
      float ho = sO[r*SW + OFF_ + j];
      float al = __expf(ls[j]) * mdt * dts;
      float v  = ho + al * (cand - ho);
      hn[j] = v; mean += v;
    }
    mean *= (1.f / H_);
    float var = 0.f;
#pragma unroll
    for (int j = 0; j < H_; ++j) { float d = hn[j] - mean; var += d*d; }
    var *= (1.f / H_);
    float rs = rsqrtf(var + 1e-5f);
#pragma unroll
    for (int j = 0; j < H_; ++j)
      sN[r*SW + OFF_ + j] = (hn[j] - mean)*rs*g[j] + be[j];
  }
}

__global__ __launch_bounds__(256) void scan_kernel(ScanArgs A) {
  __shared__ float sSt[2][16*SW];                 // ping-pong region states (f32)
  __shared__ __align__(16) f16 aBuf[16*ABW];      // WMMA A staging (f16)
  __shared__ float res[16*SW];                    // GEMM results / scratch
  __shared__ float red1[16*16], red2[16*16], mv[16*2];

  const int tid = threadIdx.x;
  const int wave = tid >> 5, lane = tid & 31;
  const int blk = blockIdx.x;                     // 8 blocks x 16 rows
  const f16* pack = A.pack;
  const float mdt = 2.f/(1.f + __expf(-A.metab[0])) + 0.5f;

  for (int i = tid; i < 2*16*SW; i += 256) ((float*)sSt)[i] = 0.f;
  __syncthreads();

  for (int t = 0; t < T_STEPS; ++t) {
    const float maskc = (t == 0) ? 0.f : 0.1f;
    float* sO = sSt[t & 1];
    float* sN = sSt[(t + 1) & 1];

    // ---- input projections: stage xt [16x8 -> pad32] ----
    for (int i = tid; i < 16*32; i += 256) {
      int r = i >> 5, c = i & 31;
      float v = (c < IN_D) ? A.x[((long)(blk*16 + r)*T_STEPS + t)*IN_D + c] : 0.f;
      aBuf[r*ABW + c] = (f16)v;
    }
    __syncthreads();
    gemm16(res, 0,  aBuf, 0, pack, 0, 1, 2, wave, lane);   // olf lin -> res[0..]
    gemm16(res, 32, aBuf, 0, pack, 2, 1, 2, wave, lane);   // opt lin -> res[32..]
    __syncthreads();

    // ---- batch A stage: olf@0(64), opt@64(64), hyp@128(96) ----
    for (int i = tid; i < 16*64; i += 256) {
      int r = i >> 6, c = i & 63; float v = 0.f;
      if (c < 20)      v = res[r*SW + c] + A.inb0[c];
      else if (c < 40) v = sO[r*SW + R_OFF[0] + (c-20)];
      aBuf[r*ABW + c] = (f16)v;
    }
    for (int i = tid; i < 16*64; i += 256) {
      int r = i >> 6, c = i & 63; float v = 0.f;
      if (c < 17)      v = res[r*SW + 32 + c] + A.inb1[c];
      else if (c < 34) v = sO[r*SW + R_OFF[1] + (c-17)];
      aBuf[r*ABW + 64 + c] = (f16)v;
    }
    for (int i = tid; i < 16*96; i += 256) {
      int r = i / 96, c = i - (i/96)*96; float v = 0.f;
      if (c < 38)      v = sO[r*SW + R_OFF[2] + c];
      else if (c < 58) v = sO[r*SW + R_OFF[7] + (c-38)];
      else if (c < 78) v = sO[r*SW + R_OFF[7] + (c-58)];
      aBuf[r*ABW + 128 + c] = (f16)v;
    }
    __syncthreads();
    gemm16(res, 0,  aBuf, 0,   pack, 4,  2, 2, wave, lane);  // olfactory
    gemm16(res, 32, aBuf, 64,  pack, 8,  2, 2, wave, lane);  // optic_nerve
    gemm16(res, 64, aBuf, 128, pack, 60, 3, 2, wave, lane);  // hypothalamus
    __syncthreads();
    region_ew<0>(0, 0,  sN, sO, res, A, mdt, tid);
    region_ew<1>(1, 32, sN, sO, res, A, mdt, tid);
    region_ew<7>(2, 64, sN, sO, res, A, mdt, tid);
    __syncthreads();

    // ---- batch B: pallium(old)@0(32) for axon, spinal zfull@32(64) ----
    for (int i = tid; i < 16*32; i += 256) {
      int r = i >> 5, c = i & 31;
      float v = (c < 25) ? sO[r*SW + R_OFF[5] + c] : 0.f;
      aBuf[r*ABW + c] = (f16)v;
    }
    for (int i = tid; i < 16*64; i += 256) {
      int r = i >> 6, c = i & 63; float v = 0.f;
      if (c < 20)      v = sN[r*SW + R_OFF[0] + c];
      else if (c < 37) v = sN[r*SW + R_OFF[1] + (c-20)];
      else if (c < 57) v = sO[r*SW + R_OFF[8] + (c-37)];
      aBuf[r*ABW + 32 + c] = (f16)v;
    }
    __syncthreads();
    gemm16(res, 0,   aBuf, 32, pack, 66, 2, 2, wave, lane);  // spinal
    gemm16(res, 64,  aBuf, 0,  pack, 78, 1, 4, wave, lane);  // p2b gate
    gemm16(res, 128, aBuf, 0,  pack, 74, 1, 4, wave, lane);  // p2b proj
    __syncthreads();
    region_ew<8>(0, 0, sN, sO, res, A, mdt, tid);
    for (int i = tid; i < 16*57; i += 256) {                  // axon -> res[192..]
      int r = i / 57, c = i - (i/57)*57;
      float gt = res[r*SW + 64  + c] + A.p2b_gb[c];
      float pj = res[r*SW + 128 + c] + A.p2b_pb[c];
      res[r*SW + 192 + c] = (1.f/(1.f+__expf(-gt))) * tanhf(pj);
    }
    __syncthreads();

    // ---- brainstem (K=133 -> 160) ----
    for (int i = tid; i < 16*160; i += 256) {
      int r = i / 160, c = i - (i/160)*160; float v = 0.f;
      if (c < 57) {
        float base;
        if (c < 20)      base = sN[r*SW + R_OFF[0] + c];
        else if (c < 37) base = sN[r*SW + R_OFF[1] + (c-20)];
        else             base = sN[r*SW + R_OFF[7] + (c-37)];
        v = base + maskc * res[r*SW + 192 + c];
      } else if (c < 95)  v = sO[r*SW + R_OFF[2] + (c-57)];
      else if (c < 133)   v = sO[r*SW + R_OFF[2] + (c-95)];
      aBuf[r*ABW + c] = (f16)v;
    }
    __syncthreads();
    gemm16(res, 0, aBuf, 0, pack, 12, 5, 3, wave, lane);
    __syncthreads();
    region_ew<2>(0, 0, sN, sO, res, A, mdt, tid);
    __syncthreads();

    // ---- tectum (93 -> 96) ----
    for (int i = tid; i < 16*96; i += 256) {
      int r = i / 96, c = i - (i/96)*96; float v = 0.f;
      if (c < 17)      v = sN[r*SW + R_OFF[1] + c];
      else if (c < 55) v = sN[r*SW + R_OFF[2] + (c-17)];
      else if (c < 93) v = sO[r*SW + R_OFF[3] + (c-55)];
      aBuf[r*ABW + c] = (f16)v;
    }
    __syncthreads();
    gemm16(res, 0, aBuf, 0, pack, 27, 3, 3, wave, lane);
    __syncthreads();
    region_ew<3>(0, 0, sN, sO, res, A, mdt, tid);
    __syncthreads();

    // ---- striatum (136 -> 160) ----
    for (int i = tid; i < 16*160; i += 256) {
      int r = i / 160, c = i - (i/160)*160; float v = 0.f;
      if (c < 38)       v = sN[r*SW + R_OFF[2] + c];
      else if (c < 76)  v = sN[r*SW + R_OFF[3] + (c-38)];
      else if (c < 106) v = sO[r*SW + R_OFF[4] + (c-76)];
      else if (c < 136) v = sO[r*SW + R_OFF[4] + (c-106)];
      aBuf[r*ABW + c] = (f16)v;
    }
    __syncthreads();
    gemm16(res, 0, aBuf, 0, pack, 36, 5, 2, wave, lane);
    __syncthreads();
    region_ew<4>(0, 0, sN, sO, res, A, mdt, tid);
    __syncthreads();

    // ---- pallium (118 -> 128) ----
    for (int i = tid; i < 16*128; i += 256) {
      int r = i >> 7, c = i & 127; float v = 0.f;
      if (c < 30)       v = sN[r*SW + R_OFF[4] + c];
      else if (c < 68)  v = sN[r*SW + R_OFF[3] + (c-30)];
      else if (c < 93)  v = sO[r*SW + R_OFF[5] + (c-68)];
      else if (c < 118) v = sO[r*SW + R_OFF[5] + (c-93)];
      aBuf[r*ABW + c] = (f16)v;
    }
    __syncthreads();
    gemm16(res, 0, aBuf, 0, pack, 46, 4, 2, wave, lane);
    __syncthreads();
    region_ew<5>(0, 0, sN, sO, res, A, mdt, tid);
    __syncthreads();

    // ---- cerebellum (88 -> 96) ----
    for (int i = tid; i < 16*96; i += 256) {
      int r = i / 96, c = i - (i/96)*96; float v = 0.f;
      if (c < 25)      v = sN[r*SW + R_OFF[5] + c];
      else if (c < 63) v = sN[r*SW + R_OFF[2] + (c-25)];
      else if (c < 88) v = sO[r*SW + R_OFF[6] + (c-63)];
      aBuf[r*ABW + c] = (f16)v;
    }
    __syncthreads();
    gemm16(res, 0, aBuf, 0, pack, 54, 3, 2, wave, lane);
    __syncthreads();
    region_ew<6>(0, 0, sN, sO, res, A, mdt, tid);
    __syncthreads();

    // ---- motor (62 -> 64) ----
    for (int i = tid; i < 16*64; i += 256) {
      int r = i >> 6, c = i & 63; float v = 0.f;
      if (c < 25)      v = sN[r*SW + R_OFF[6] + c];
      else if (c < 45) v = sN[r*SW + R_OFF[8] + (c-25)];
      else if (c < 62) v = sO[r*SW + R_OFF[9] + (c-45)];
      aBuf[r*ABW + c] = (f16)v;
    }
    __syncthreads();
    gemm16(res, 0, aBuf, 0, pack, 70, 2, 2, wave, lane);
    __syncthreads();
    region_ew<9>(0, 0, sN, sO, res, A, mdt, tid);
    __syncthreads();

    // ---- thalamus: stage allr (250 -> 256) ----
    for (int i = tid; i < 16*256; i += 256) {
      int r = i >> 8, c = i & 255;
      aBuf[r*ABW + c] = (f16)((c < 250) ? sN[r*SW + c] : 0.f);
    }
    __syncthreads();

    // A fragments are reused across all 16 N-tiles: preload once per wave.
    v16h aT[8];
#pragma unroll
    for (int kt = 0; kt < 8; ++kt) aT[kt] = load_a(aBuf, kt*32, lane);

    // GEMM 1: gate pre-activation
    for (int nt = wave; nt < 16; nt += 8) {
      v8f acc = {};
#pragma unroll
      for (int kt = 0; kt < 8; ++kt) {
        v16h b = load_b(pack, 82 + nt*8 + kt, lane);
        acc = __builtin_amdgcn_wmma_f32_16x16x32_f16(false, aT[kt], false, b, (short)0, acc, false, false);
      }
      int n = nt*16 + (lane & 15), mh = (lane >> 4) * 8;
#pragma unroll
      for (int r = 0; r < 8; ++r) res[(mh + r)*SW + n] = acc[r];
    }
    __syncthreads();

    // LN + sigmoid gate (row-parallel 3-phase reduction)
    {
      int r = tid >> 4, c = tid & 15;
      float s = 0.f, s2 = 0.f;
      for (int j = c*16; j < c*16 + 16; ++j) {
        float v = res[r*SW + j] + A.th_b[j]; s += v; s2 += v*v;
      }
      red1[r*16 + c] = s; red2[r*16 + c] = s2;
    }
    __syncthreads();
    if (tid < 16) {
      float s = 0.f, s2 = 0.f;
      for (int k = 0; k < 16; ++k) { s += red1[tid*16 + k]; s2 += red2[tid*16 + k]; }
      float m = s / 256.f;
      mv[tid*2] = m; mv[tid*2 + 1] = rsqrtf(s2/256.f - m*m + 1e-5f);
    }
    __syncthreads();
    {
      int r = tid >> 4, c = tid & 15;
      float m = mv[r*2], rs = mv[r*2 + 1];
      for (int j = c*16; j < c*16 + 16; ++j) {
        float v  = res[r*SW + j] + A.th_b[j];
        float ln = (v - m)*rs*A.th_g[j] + A.th_b2[j];
        res[r*SW + j] = 1.f/(1.f + __expf(-ln));
      }
    }
    __syncthreads();

    // GEMM 2 (thalamic_proj) with fused gate*(acc+b) epilogue -> global hidden
    for (int nt = wave; nt < 16; nt += 8) {
      v8f acc = {};
#pragma unroll
      for (int kt = 0; kt < 8; ++kt) {
        v16h b = load_b(pack, 210 + nt*8 + kt, lane);
        acc = __builtin_amdgcn_wmma_f32_16x16x32_f16(false, aT[kt], false, b, (short)0, acc, false, false);
      }
      int n = nt*16 + (lane & 15), mh = (lane >> 4) * 8;
#pragma unroll
      for (int r = 0; r < 8; ++r) {
        int m = mh + r;
        float o = res[m*SW + n] * (acc[r] + A.tp_b[n]);
        A.hid[((long)(blk*16 + m)*T_STEPS + t)*HID + n] = o;
      }
    }
    __syncthreads();
  }
}

// ---------------- post-processing: preds / anomaly / surprise ------------
__device__ __forceinline__ float wred(float v) {
  for (int m = 16; m > 0; m >>= 1) v += __shfl_xor(v, m, 32);
  return v;
}

__global__ __launch_bounds__(256) void post_kernel(
    const float* hid, const float* x,
    const float* ng, const float* nb, const float* nhW, const float* nhB,
    const float* ag, const float* ab, const float* aW, const float* aB,
    float* preds, float* anom, float* surp) {
  int wave = threadIdx.x >> 5, lane = threadIdx.x & 31;
  long row = (long)blockIdx.x*8 + wave;
  if (row >= (long)BATCH*T_STEPS) return;
  int b = (int)(row / T_STEPS), t = (int)(row - (long)b*T_STEPS);
  const float* hr = hid + row*HID;
  float v[8]; float s = 0.f, s2 = 0.f;
#pragma unroll
  for (int i = 0; i < 8; ++i) { v[i] = hr[lane + 32*i]; s += v[i]; s2 += v[i]*v[i]; }
  s = wred(s); s2 = wred(s2);
  float mean = s/256.f, rs = rsqrtf(s2/256.f - mean*mean + 1e-5f);
  // anomaly
  float da = 0.f;
#pragma unroll
  for (int i = 0; i < 8; ++i) {
    int k = lane + 32*i;
    da += ((v[i]-mean)*rs*ag[k] + ab[k]) * aW[k];
  }
  da = wred(da);
  if (lane == 0) anom[row] = 1.f/(1.f + __expf(-(da + aB[0])));
  // preds + surprise
  if (t < T_STEPS - 1) {
    float serr = 0.f;
    for (int o = 0; o < IN_D; ++o) {
      float d = 0.f;
#pragma unroll
      for (int i = 0; i < 8; ++i) {
        int k = lane + 32*i;
        d += ((v[i]-mean)*rs*ng[k] + nb[k]) * nhW[o*HID + k];
      }
      d = wred(d);
      if (lane == 0) {
        float pred = d + nhB[o];
        preds[((long)b*(T_STEPS-1) + t)*IN_D + o] = pred;
        float e = pred - x[((long)b*T_STEPS + (t+1))*IN_D + o];
        serr += e*e;
      }
    }
    if (lane == 0) surp[(long)b*(T_STEPS-1) + t] = serr / IN_D;
  }
}

// ---------------- props head (last timestep) -----------------------------
__global__ __launch_bounds__(128) void props_kernel(
    const float* hid, const float* pg, const float* pb,
    const float* w1, const float* b1, const float* w2, const float* b2, float* out) {
  int r = blockIdx.x, tid = threadIdx.x;  // 128 threads
  const float* hr = hid + ((long)r*T_STEPS + (T_STEPS-1))*HID;
  __shared__ float rb[128]; __shared__ float lnv[HID]; __shared__ float mid[128];
  __shared__ float mvp[2];
  float a = hr[tid], c = hr[tid + 128];
  rb[tid] = a + c; __syncthreads();
  for (int s = 64; s > 0; s >>= 1) { if (tid < s) rb[tid] += rb[tid + s]; __syncthreads(); }
  if (tid == 0) mvp[0] = rb[0] / 256.f;
  __syncthreads();
  rb[tid] = a*a + c*c; __syncthreads();
  for (int s = 64; s > 0; s >>= 1) { if (tid < s) rb[tid] += rb[tid + s]; __syncthreads(); }
  if (tid == 0) { float m = mvp[0]; mvp[1] = rsqrtf(rb[0]/256.f - m*m + 1e-5f); }
  __syncthreads();
  float m = mvp[0], rs = mvp[1];
  lnv[tid]       = (a - m)*rs*pg[tid]       + pb[tid];
  lnv[tid + 128] = (c - m)*rs*pg[tid + 128] + pb[tid + 128];
  __syncthreads();
  float d = b1[tid];
  for (int k = 0; k < HID; ++k) d += lnv[k] * w1[tid*HID + k];
  mid[tid] = 0.5f * d * (1.f + erff(d * 0.70710678118f));   // exact gelu
  __syncthreads();
  if (tid < 2) {
    float o = b2[tid];
    for (int k = 0; k < 128; ++k) o += mid[k] * w2[tid*128 + k];
    out[r*2 + tid] = o;
  }
}

// ---------------- host launcher ------------------------------------------
extern "C" void kernel_launch(void* const* d_in, const int* in_sizes, int n_in,
                              void* d_out, int out_size, void* d_ws, size_t ws_size,
                              hipStream_t stream) {
  (void)in_sizes; (void)n_in; (void)out_size; (void)ws_size;
  auto F = [&](int i) { return (const float*)d_in[i]; };
  float* out = (float*)d_out;
  f16* pack = (f16*)d_ws;

  // d_in layout: 0=x, 1=metabolic, 2/3=input_proj W/b, 4/5=optic W/b,
  // 6+7i.. = region {Wi,bi,Wh,bh,log_step,g,beta}, 76..79 p2b, 80..85 thalamus,
  // 86..99 heads.
  PackArgs pa{};
  auto setM = [&](int idx, const float* W0, const float* W1, int K0, int K, int N, int base) {
    pa.m[idx] = { W0, W1, K0, K, N, base, (K + 31)/32, (N + 15)/16 };
  };
  constexpr int RFB[10] = {4,8,12,27,36,46,54,60,66,70};
  constexpr int RK0[10] = {20,17,95,55,106,93,63,58,37,45};
  constexpr int RK [10] = {40,34,133,93,136,118,88,78,57,62};
  setM(0, F(2), nullptr, 8, 8, 20, 0);
  setM(1, F(4), nullptr, 8, 8, 17, 2);
  for (int i = 0; i < 10; ++i)
    setM(2 + i, F(6 + 7*i + 0), F(6 + 7*i + 2), RK0[i], RK[i], R_H[i], RFB[i]);
  setM(12, F(76), nullptr, 25, 25, 57, 74);   // p2b proj
  setM(13, F(78), nullptr, 25, 25, 57, 78);   // p2b gate
  setM(14, F(80), nullptr, 250, 250, 256, 82);  // thalamus
  setM(15, F(84), nullptr, 250, 250, 256, 210); // thalamic_proj

  pack_kernel<<<NFRAGS, 512, 0, stream>>>(pa, pack);

  ScanArgs sa{};
  sa.x = F(0); sa.metab = F(1); sa.inb0 = F(3); sa.inb1 = F(5);
  for (int i = 0; i < 10; ++i) {
    int b = 6 + 7*i;
    sa.bi[i] = F(b+1); sa.bh[i] = F(b+3); sa.ls[i] = F(b+4);
    sa.g[i]  = F(b+5); sa.be[i] = F(b+6);
  }
  sa.p2b_pb = F(77); sa.p2b_gb = F(79);
  sa.th_b = F(81); sa.th_g = F(82); sa.th_b2 = F(83); sa.tp_b = F(85);
  sa.pack = pack; sa.hid = out + O_HIDDEN;

  scan_kernel<<<8, 256, 0, stream>>>(sa);

  post_kernel<<<(BATCH*T_STEPS)/8, 256, 0, stream>>>(
      out + O_HIDDEN, F(0),
      F(86), F(87), F(88), F(89),
      F(90), F(91), F(92), F(93),
      out, out + O_ANOM, out + O_SURP);

  props_kernel<<<BATCH, 128, 0, stream>>>(
      out + O_HIDDEN, F(94), F(95), F(96), F(97), F(98), F(99), out + O_PROPS);
}
